// Model_28243704939364
// MI455X (gfx1250) — matrix operand order repaired and checked by reading, not compiled, not run on hardware
//
#include <hip/hip_runtime.h>

// ---------------- problem constants (from reference setup_inputs) ------------
#define BDIM   2
#define SDIM   2048
#define HDIM   2048
#define EDIM   28
#define MDIM   1408
#define TOPK   6
#define TTOK   (BDIM * SDIM)          // 4096 tokens

typedef __bf16 bf16_t;
typedef bf16_t v16bf __attribute__((ext_vector_type(16)));
typedef float  v8f   __attribute__((ext_vector_type(8)));
typedef int    int4v __attribute__((ext_vector_type(4)));

typedef __attribute__((address_space(1))) int4v gas_i4;   // global int4*
typedef __attribute__((address_space(3))) int4v lds_i4;   // LDS int4*

// ---------------------------------------------------------------------------
// helpers
// ---------------------------------------------------------------------------

// async 16B copy global(bf16, already converted) -> LDS; falls back to a
// register copy if the gfx1250 async builtin is unavailable.
__device__ __forceinline__ void async_cp16(const bf16_t* g, bf16_t* l) {
#if __has_builtin(__builtin_amdgcn_global_load_async_to_lds_b128)
    gas_i4* gp = (gas_i4*)(unsigned long long)(g);
    lds_i4* lp = (lds_i4*)(unsigned)(unsigned long long)(l);
    __builtin_amdgcn_global_load_async_to_lds_b128(gp, lp, 0, 0);
#else
    *(uint4*)l = *(const uint4*)g;
#endif
}

__device__ __forceinline__ void wait_async0() {
#if __has_builtin(__builtin_amdgcn_global_load_async_to_lds_b128)
#if __has_builtin(__builtin_amdgcn_s_wait_asynccnt)
    __builtin_amdgcn_s_wait_asynccnt(0);
#else
    asm volatile("s_wait_asynccnt 0" ::: "memory");
#endif
#endif
}

// two contiguous b128 LDS loads -> one 16xbf16 WMMA fragment
__device__ __forceinline__ v16bf load_frag(const bf16_t* p) {
    union { uint4 u[2]; v16bf v; } t;
    t.u[0] = *(const uint4*)(p);
    t.u[1] = *(const uint4*)(p + 8);
    return t.v;
}

// gather 8 strided fp32, convert to bf16, pack into 16B
__device__ __forceinline__ uint4 pack8(const float* s, size_t stride) {
    union { bf16_t h[8]; uint4 u; } t;
    #pragma unroll
    for (int j = 0; j < 8; j++) t.h[j] = (bf16_t)s[j * stride];
    return t.u;
}

// fast silu: x * rcp(1+exp(-x)) — v_rcp_f32 instead of the IEEE div sequence
__device__ __forceinline__ float fast_silu(float x) {
    return x * __builtin_amdgcn_rcpf(1.0f + __expf(-x));
}

// ---------------------------------------------------------------------------
// small utility kernels
// ---------------------------------------------------------------------------
__global__ void zero_f32_kernel(float* __restrict__ p, int n) {
    int i = blockIdx.x * blockDim.x + threadIdx.x;
    if (i < n) p[i] = 0.0f;
}

__global__ void zero_i32_kernel(int* __restrict__ p, int n) {
    int i = blockIdx.x * blockDim.x + threadIdx.x;
    if (i < n) p[i] = 0;
}

__global__ void cvt_bf16_kernel(const float* __restrict__ x,
                                bf16_t* __restrict__ xb, int n) {
    int i = blockIdx.x * blockDim.x + threadIdx.x;
    if (i < n) xb[i] = (bf16_t)x[i];
}

// ---------------------------------------------------------------------------
// gating: logits = x @ gate_weight^T ; softmax ; top-6 ; normalize
// ---------------------------------------------------------------------------
__global__ __launch_bounds__(256)
void gate_kernel(const float* __restrict__ x,       // [T,H]
                 const float* __restrict__ gw,      // [E,H]
                 int*   __restrict__ topk_idx,      // [T,K]
                 float* __restrict__ topk_w,        // [T,K]
                 int*   __restrict__ counts)        // [E]
{
    const int t   = blockIdx.x;
    const int tid = threadIdx.x;
    __shared__ float slog[32];

    const int e   = tid >> 3;
    const int sub = tid & 7;
    float acc = 0.0f;
    if (e < EDIM) {
        const float* xr = x  + (size_t)t * HDIM;
        const float* gr = gw + (size_t)e * HDIM;
        for (int h = sub; h < HDIM; h += 8)
            acc += xr[h] * gr[h];
    }
    acc += __shfl_down(acc, 4, 8);
    acc += __shfl_down(acc, 2, 8);
    acc += __shfl_down(acc, 1, 8);
    if (sub == 0 && e < EDIM) slog[e] = acc;
    __syncthreads();

    if (tid == 0) {
        float mx = -1e30f;
        for (int i = 0; i < EDIM; i++) mx = fmaxf(mx, slog[i]);
        float sc[EDIM];
        float sum = 0.0f;
        for (int i = 0; i < EDIM; i++) { sc[i] = __expf(slog[i] - mx); sum += sc[i]; }
        float inv = 1.0f / sum;
        for (int i = 0; i < EDIM; i++) sc[i] *= inv;

        int   bi[TOPK];
        float bw[TOPK];
        float wsum = 0.0f;
        for (int k = 0; k < TOPK; k++) {
            float best = -1.0f; int bidx = 0;
            for (int i = 0; i < EDIM; i++)
                if (sc[i] > best) { best = sc[i]; bidx = i; }
            bi[k] = bidx; bw[k] = best; sc[bidx] = -2.0f; wsum += best;
        }
        float nw = 1.0f / (wsum + 1e-20f);
        for (int k = 0; k < TOPK; k++) {
            topk_idx[t * TOPK + k] = bi[k];
            topk_w  [t * TOPK + k] = bw[k] * nw;
            atomicAdd(&counts[bi[k]], 1);
        }
    }
}

// exclusive scan over 28 expert counts (trivial)
__global__ void scan_kernel(const int* __restrict__ counts,
                            int* __restrict__ offsets) {
    if (threadIdx.x == 0 && blockIdx.x == 0) {
        int s = 0;
        for (int e = 0; e < EDIM; e++) { offsets[e] = s; s += counts[e]; }
        offsets[EDIM] = s;
    }
}

// deterministic token-list build: one wave32 per expert, ballot-prefix append
__global__ __launch_bounds__(32)
void build_lists_kernel(const int*   __restrict__ topk_idx,
                        const float* __restrict__ topk_w,
                        const int*   __restrict__ offsets,
                        int*   __restrict__ perm,      // [T*K] token id per slot
                        float* __restrict__ slotw)     // [T*K] gate weight per slot
{
    const int e    = blockIdx.x;
    const int lane = threadIdx.x;
    const int base = offsets[e];
    int run = 0;
    for (int t0 = 0; t0 < TTOK; t0 += 32) {
        const int t = t0 + lane;
        bool  sel = false;
        float w   = 0.0f;
        for (int k = 0; k < TOPK; k++) {
            if (topk_idx[t * TOPK + k] == e) { sel = true; w = topk_w[t * TOPK + k]; }
        }
        unsigned long long m = __ballot(sel);
        int pos = __popcll(m & ((1ull << lane) - 1ull));
        if (sel) {
            perm [base + run + pos] = t;
            slotw[base + run + pos] = w;
        }
        run += __popcll(m);
    }
}

// ---------------------------------------------------------------------------
// LDS layouts (fragment-major):
//  A tile: lA[row][48], row stored k-reordered [k0..7 | k16..23 | k8..15 | k24..31]
//          -> lane reads &lA[row][hi*16], 32 contiguous bytes (2 x b128)
//  B tile: lB[col][48] column-major -> lane reads &lB[col][hi*16], contiguous
// ---------------------------------------------------------------------------

// Expert GEMM 1 (fused): h = silu(x @ wg) * (x @ wu)     [cnt, M] bf16
__global__ __launch_bounds__(256)
void mlp_up_kernel(const bf16_t* __restrict__ xb,     // [T,H] bf16
                   const float*  __restrict__ wg,     // [H,M] expert slice fp32
                   const float*  __restrict__ wu,     // [H,M]
                   const int*    __restrict__ perm,
                   const int*    __restrict__ offsets,
                   int e,
                   bf16_t* __restrict__ hbuf)         // [T,M] (local rows)
{
    const int off  = offsets[e];
    const int cnt  = offsets[e + 1] - off;
    const int row0 = blockIdx.x * 128;
    if (row0 >= cnt) return;
    const int col0 = blockIdx.y * 64;

    __shared__ int    ltok[128];
    __shared__ bf16_t lX[128][48];   // padded rows, 16B-aligned chunks
    __shared__ bf16_t lG[64][48];    // column-major
    __shared__ bf16_t lU[64][48];

    const int tid = threadIdx.x;
    if (tid < 128) {
        int r = row0 + tid;
        ltok[tid] = perm[off + (r < cnt ? r : 0)];
    }
    __syncthreads();

    const int lane = tid & 31, wave = tid >> 5;
    const int l16 = lane & 15, hi = lane >> 4;
    const int wr = wave * 16;

    v8f accG[4], accU[4];
    #pragma unroll
    for (int ct = 0; ct < 4; ct++)
        #pragma unroll
        for (int i = 0; i < 8; i++) { accG[ct][i] = 0.0f; accU[ct][i] = 0.0f; }

    const int xrow = tid >> 1, xhalf = tid & 1;  // A-tile loader: 2 thr/row
    const int wc2 = tid & 63, wk2 = tid >> 6;    // B-tile loader: col, k-octet

    for (int k0 = 0; k0 < HDIM; k0 += 32) {
        // ---- A tile: async bf16 copy, k-reordered chunk placement ----
        {
            const bf16_t* gsrc = xb + (size_t)ltok[xrow] * HDIM + k0 + xhalf * 16;
            bf16_t* d0 = &lX[xrow][xhalf ? 8 : 0];
            bf16_t* d1 = &lX[xrow][xhalf ? 24 : 16];
            async_cp16(gsrc,     d0);
            async_cp16(gsrc + 8, d1);
        }
        // ---- B tiles: coalesced fp32 column loads -> packed b128 store ----
        {
            const float* gcol = wg + (size_t)(k0 + wk2 * 8) * MDIM + col0 + wc2;
            const float* ucol = wu + (size_t)(k0 + wk2 * 8) * MDIM + col0 + wc2;
            *(uint4*)&lG[wc2][(wk2 & 1) * 16 + (wk2 >> 1) * 8] = pack8(gcol, MDIM);
            *(uint4*)&lU[wc2][(wk2 & 1) * 16 + (wk2 >> 1) * 8] = pack8(ucol, MDIM);
            __builtin_prefetch(gcol + (size_t)32 * MDIM, 0, 1);
            __builtin_prefetch(ucol + (size_t)32 * MDIM, 0, 1);
        }
        wait_async0();
        __syncthreads();

        const v16bf af = load_frag(&lX[wr + l16][hi * 16]);
        #pragma unroll
        for (int ct = 0; ct < 4; ct++) {
            const int bc = ct * 16 + l16;
            const v16bf bg = load_frag(&lG[bc][hi * 16]);
            const v16bf bu = load_frag(&lU[bc][hi * 16]);
            accG[ct] = __builtin_amdgcn_wmma_f32_16x16x32_bf16(
                false, af, false, bg, (short)0, accG[ct], false, false);
            accU[ct] = __builtin_amdgcn_wmma_f32_16x16x32_bf16(
                false, af, false, bu, (short)0, accU[ct], false, false);
        }
        __syncthreads();
    }

    // epilogue: silu(g) * u -> bf16 hbuf
    #pragma unroll
    for (int ct = 0; ct < 4; ct++) {
        #pragma unroll
        for (int r = 0; r < 8; r++) {
            const int lr = row0 + wr + hi * 8 + r;
            if (lr < cnt) {
                float h = fast_silu(accG[ct][r]) * accU[ct][r];
                hbuf[(size_t)lr * MDIM + col0 + ct * 16 + l16] = (bf16_t)h;
            }
        }
    }
}

// Expert GEMM 2: y[token] += slot_w * (h @ wd)          K = M = 1408
__global__ __launch_bounds__(256)
void mlp_down_kernel(const bf16_t* __restrict__ hbuf,   // [T,M] local rows
                     const float*  __restrict__ wd,     // [M,H] expert slice
                     const int*    __restrict__ perm,
                     const float*  __restrict__ slotw,
                     const int*    __restrict__ offsets,
                     int e,
                     float* __restrict__ y)             // [T,H]
{
    const int off  = offsets[e];
    const int cnt  = offsets[e + 1] - off;
    const int row0 = blockIdx.x * 128;
    if (row0 >= cnt) return;
    const int col0 = blockIdx.y * 64;

    __shared__ int    ltok[128];
    __shared__ float  lw[128];
    __shared__ bf16_t lA[128][48];
    __shared__ bf16_t lB[64][48];    // column-major

    const int tid = threadIdx.x;
    if (tid < 128) {
        int r = row0 + tid;
        int rr = r < cnt ? r : 0;
        ltok[tid] = perm [off + rr];
        lw  [tid] = slotw[off + rr];
    }
    __syncthreads();

    const int lane = tid & 31, wave = tid >> 5;
    const int l16 = lane & 15, hi = lane >> 4;
    const int wr = wave * 16;

    v8f acc[4];
    #pragma unroll
    for (int ct = 0; ct < 4; ct++)
        #pragma unroll
        for (int i = 0; i < 8; i++) acc[ct][i] = 0.0f;

    const int arow = tid >> 1, ahalf = tid & 1;
    const int wc2 = tid & 63, wk2 = tid >> 6;

    for (int k0 = 0; k0 < MDIM; k0 += 32) {
        // ---- A tile from hbuf (bf16): async, k-reordered ----
        {
            int lr = row0 + arow;
            int rr = lr < cnt ? lr : 0;
            const bf16_t* gsrc = hbuf + (size_t)rr * MDIM + k0 + ahalf * 16;
            bf16_t* d0 = &lA[arow][ahalf ? 8 : 0];
            bf16_t* d1 = &lA[arow][ahalf ? 24 : 16];
            async_cp16(gsrc,     d0);
            async_cp16(gsrc + 8, d1);
        }
        // ---- B tile: coalesced fp32 column loads -> packed b128 store ----
        {
            const float* dcol = wd + (size_t)(k0 + wk2 * 8) * HDIM + col0 + wc2;
            *(uint4*)&lB[wc2][(wk2 & 1) * 16 + (wk2 >> 1) * 8] = pack8(dcol, HDIM);
            __builtin_prefetch(dcol + (size_t)32 * HDIM, 0, 1);
        }
        wait_async0();
        __syncthreads();

        const v16bf af = load_frag(&lA[wr + l16][hi * 16]);
        #pragma unroll
        for (int ct = 0; ct < 4; ct++) {
            const v16bf bfr = load_frag(&lB[ct * 16 + l16][hi * 16]);
            acc[ct] = __builtin_amdgcn_wmma_f32_16x16x32_bf16(
                false, af, false, bfr, (short)0, acc[ct], false, false);
        }
        __syncthreads();
    }

    // weighted scatter-add into y (one writer per element within this kernel;
    // experts run sequentially on the stream -> deterministic)
    #pragma unroll
    for (int ct = 0; ct < 4; ct++) {
        #pragma unroll
        for (int r = 0; r < 8; r++) {
            const int lr = row0 + wr + hi * 8 + r;
            if (lr < cnt) {
                const int   tok = ltok[wr + hi * 8 + r];
                const float w   = lw  [wr + hi * 8 + r];
                float* yp = y + (size_t)tok * HDIM + col0 + ct * 16 + l16;
                *yp += w * acc[ct][r];
            }
        }
    }
}

// ---------------------------------------------------------------------------
// launcher
// ---------------------------------------------------------------------------
extern "C" void kernel_launch(void* const* d_in, const int* in_sizes, int n_in,
                              void* d_out, int out_size, void* d_ws, size_t ws_size,
                              hipStream_t stream) {
    const float* x      = (const float*)d_in[0];   // [B,S,H]
    const float* gw     = (const float*)d_in[1];   // [E,H]
    const float* w_gate = (const float*)d_in[2];   // [E,H,M]
    const float* w_up   = (const float*)d_in[3];   // [E,H,M]
    const float* w_down = (const float*)d_in[4];   // [E,M,H]
    float* y = (float*)d_out;                      // [B,S,H] fp32

    // workspace carve-out (256B aligned)
    char* ws = (char*)d_ws;
    size_t p = 0;
    auto carve = [&](size_t bytes) -> void* {
        p = (p + 255) & ~(size_t)255;
        void* q = ws + p;
        p += bytes;
        return q;
    };
    int*    counts   = (int*)   carve(EDIM * sizeof(int));
    int*    offsets  = (int*)   carve((EDIM + 1) * sizeof(int));
    int*    topk_idx = (int*)   carve((size_t)TTOK * TOPK * sizeof(int));
    float*  topk_w   = (float*) carve((size_t)TTOK * TOPK * sizeof(float));
    int*    perm     = (int*)   carve((size_t)TTOK * TOPK * sizeof(int));
    float*  slotw    = (float*) carve((size_t)TTOK * TOPK * sizeof(float));
    bf16_t* xb       = (bf16_t*)carve((size_t)TTOK * HDIM * sizeof(bf16_t));
    bf16_t* hbuf     = (bf16_t*)carve((size_t)TTOK * MDIM * sizeof(bf16_t));
    (void)ws_size; (void)in_sizes; (void)n_in; (void)out_size;

    const int nTH = TTOK * HDIM;

    zero_i32_kernel<<<1, 64, 0, stream>>>(counts, EDIM);
    zero_f32_kernel<<<(nTH + 255) / 256, 256, 0, stream>>>(y, nTH);
    cvt_bf16_kernel<<<(nTH + 255) / 256, 256, 0, stream>>>(x, xb, nTH);

    gate_kernel<<<TTOK, 256, 0, stream>>>(x, gw, topk_idx, topk_w, counts);
    scan_kernel<<<1, 1, 0, stream>>>(counts, offsets);
    build_lists_kernel<<<EDIM, 32, 0, stream>>>(topk_idx, topk_w, offsets, perm, slotw);

    const dim3 gridUp (TTOK / 128, MDIM / 64);   // 32 x 22
    const dim3 gridDn (TTOK / 128, HDIM / 64);   // 32 x 32
    for (int e = 0; e < EDIM; e++) {
        const float* wgp = w_gate + (size_t)e * HDIM * MDIM;
        const float* wup = w_up   + (size_t)e * HDIM * MDIM;
        const float* wdp = w_down + (size_t)e * MDIM * HDIM;
        mlp_up_kernel  <<<gridUp, 256, 0, stream>>>(xb, wgp, wup, perm, offsets, e, hbuf);
        mlp_down_kernel<<<gridDn, 256, 0, stream>>>(hbuf, wdp, perm, slotw, offsets, e, y);
    }
}
